// SparseCrossAttention_49469433315809
// MI455X (gfx1250) — matrix operand order repaired and checked by reading, not compiled, not run on hardware
//
#include <hip/hip_runtime.h>

// ---------------- problem constants (fixed by the reference) ----------------
#define DIM    256    // D
#define KVDIM  512    // KV
#define MKEYS  1024   // M
#define NBATCH 4      // B
#define NHEAD  8      // H
#define DH     32     // D / H  (== WMMA K for bf16 -> one WMMA per score tile)
#define CHUNK  128    // keys per flash chunk
#define KROWPAD 40    // staged K/V row stride in bf16 (80B = 5*16B, bank-friendly)
#define SCALE  0.17677669529663687f   // 1/sqrt(32)

// ---------------- WMMA fragment types / layout helpers ---------------------
typedef __attribute__((ext_vector_type(16))) __bf16 v16bf;
typedef __attribute__((ext_vector_type(8)))  float  v8f;

union BF16x16 {
    v16bf    v;
    __bf16   h[16];   // h[2s], h[2s+1] = low/high 16 bits of VGPR slot s
    unsigned u[8];    // packed pair loads (consecutive even K)
};

// A-matrix (16x32 bf16, MxK): slot s of lane (half = lane/16) holds K = a_kbase, +1
__device__ __forceinline__ int a_kbase(int s, int half) {
    return (s >> 2) * 16 + half * 8 + (s & 3) * 2;
}
// B-matrix (32x16 bf16, KxN): slot s holds K = b_kbase, b_kbase+1 ; N = lane%16
__device__ __forceinline__ int b_kbase(int s, int half) {
    return (s >> 2) * 16 + half * 8 + (s & 3) * 2;
}

__device__ __forceinline__ v8f wmma_bf16(const BF16x16& A, const BF16x16& B, v8f C) {
    return __builtin_amdgcn_wmma_f32_16x16x32_bf16(
        false, A.v, false, B.v, (short)0, C, false, false);
}

__device__ __forceinline__ v8f zero8() {
    v8f z = {0.f, 0.f, 0.f, 0.f, 0.f, 0.f, 0.f, 0.f};
    return z;
}

// ---- async global->LDS staging of a [CHUNK x 32]-bf16 slice (row stride DIM)
// 512 b128 transfers, 16 per lane, via GLOBAL_LOAD_ASYNC_TO_LDS_B128 (ASYNCcnt)
__device__ __forceinline__ void stage_async_chunk(
    const __bf16* __restrict__ g0,   // &src[chunk_row0 * DIM + dcol0]
    __bf16 (*dst)[KROWPAD], int lane)
{
#pragma unroll
    for (int t = 0; t < 16; ++t) {
        const int idx = t * 32 + lane;     // 0..511
        const int row = idx >> 2;          // 0..127
        const int q   = idx & 3;           // 16B quarter of the 64B row
        const unsigned lds =
            (unsigned)(size_t)(const void*)&dst[row][0] + (unsigned)(q * 16);
        const unsigned long long ga =
            (unsigned long long)(size_t)(const void*)(g0 + (long)row * DIM)
            + (unsigned long long)(q * 16);
        asm volatile("global_load_async_to_lds_b128 %0, %1, off"
                     :: "v"(lds), "v"(ga) : "memory");
    }
}

// ============================================================================
// Kernel 0: convert Wq / Wo to bf16 once (packed-pair B-fragment loads later)
// ============================================================================
__global__ void wconv_kernel(const float* __restrict__ Wq, const float* __restrict__ Wo,
                             __bf16* __restrict__ Wqb, __bf16* __restrict__ Wob)
{
    const int i = blockIdx.x * blockDim.x + threadIdx.x;
    if (i < DIM * DIM) {
        Wqb[i] = (__bf16)Wq[i];
        Wob[i] = (__bf16)Wo[i];
    }
}

// ============================================================================
// Kernel 1: K = y @ Wk^T + bk, V = y @ Wv^T + bv   ->  bf16 in workspace
//   grid: (B*M/16, D/16, 2[K|V]), block: 32 (one wave)
// ============================================================================
__global__ __launch_bounds__(32) void kv_proj_kernel(
    const float* __restrict__ y,
    const float* __restrict__ Wk, const float* __restrict__ bk,
    const float* __restrict__ Wv, const float* __restrict__ bv,
    __bf16* __restrict__ Kw, __bf16* __restrict__ Vw)
{
    const int lane = threadIdx.x;
    const int half = lane >> 4, ln = lane & 15;
    const int mtile = blockIdx.x;
    const int nt    = blockIdx.y;
    const int z     = blockIdx.z;

    const float* W    = z ? Wv : Wk;
    const float* bias = z ? bv : bk;
    __bf16*      dst  = z ? Vw : Kw;

    const int b  = mtile / (MKEYS / 16);
    const int mt = mtile % (MKEYS / 16);
    const float* yb = y + (long)b * MKEYS * KVDIM;

    const int mrowA = mt * 16 + ln;
    const int ncol  = nt * 16 + ln;

    v8f acc = zero8();
    for (int kc = 0; kc < KVDIM; kc += 32) {
        BF16x16 A, Bf;
#pragma unroll
        for (int s = 0; s < 8; ++s) {
            const int ka = kc + a_kbase(s, half);
            A.h[2*s]   = (__bf16)yb[(long)mrowA * KVDIM + ka];
            A.h[2*s+1] = (__bf16)yb[(long)mrowA * KVDIM + ka + 1];
            const int kb = kc + b_kbase(s, half);
            const float* wr = W + (long)ncol * KVDIM + kb;
            Bf.h[2*s]   = (__bf16)wr[0];
            Bf.h[2*s+1] = (__bf16)wr[1];
        }
        acc = wmma_bf16(A, Bf, acc);
    }

    const float bv_ = bias[ncol];
#pragma unroll
    for (int r = 0; r < 8; ++r) {
        const int mrow = mt * 16 + half * 8 + r;
        dst[((long)b * MKEYS + mrow) * DIM + ncol] = (__bf16)(acc[r] + bv_);
    }
}

// ============================================================================
// Kernel 2: per 16-row query tile: LN -> Q proj (WMMA) -> per-head flash
//           attention with async-DMA double-buffered K/V chunks (WMMA) ->
//           Wo proj (WMMA) -> residual -> LN -> store. One wave per tile.
// ============================================================================
__global__ __launch_bounds__(32) void attn_kernel(
    const float* __restrict__ F,  const int* __restrict__ bidx,
    const __bf16* __restrict__ Wqb, const float* __restrict__ bq,
    const __bf16* __restrict__ Wob, const float* __restrict__ bo,
    const float* __restrict__ ln_g, const float* __restrict__ ln_b,
    const __bf16* __restrict__ Kw, const __bf16* __restrict__ Vw,
    float* __restrict__ out, int nrows)
{
    __shared__ float  sQin[16][DIM + 2];     // LN(F) tile (residual source)
    __shared__ __bf16 sQ  [16][DIM + 4];     // Q projection, bf16
    __shared__ float  sS  [16][CHUNK + 2];   // score chunk
    __shared__ __bf16 sP  [16][CHUNK + 4];   // softmax probs chunk, bf16
    __shared__ float  sO  [16][DIM + 2];     // attention output accumulator
    __shared__ float  sR  [16][DIM + 2];     // residual + attn (pre final LN)
    __shared__ __attribute__((aligned(16))) __bf16 sK[2][CHUNK][KROWPAD]; // async-staged K slice
    __shared__ __attribute__((aligned(16))) __bf16 sV[2][CHUNK][KROWPAD]; // async-staged V slice
    __shared__ int    sBidx[16];
    __shared__ int    sRange[2];

    const int lane = threadIdx.x;
    const int half = lane >> 4, ln = lane & 15;
    const int r0 = blockIdx.x * 16;

    // ---- load F rows + first LayerNorm (one lane per row) ----
    if (lane < 16) {
        const int gr = r0 + lane;
        int bv = -1;
        if (gr < nrows) {
            const float* frow = F + (long)gr * DIM;
            float mu = 0.f;
            for (int i = 0; i < DIM; ++i) mu += frow[i];
            mu *= (1.f / DIM);
            float var = 0.f;
            for (int i = 0; i < DIM; ++i) { float d = frow[i] - mu; var += d * d; }
            var *= (1.f / DIM);
            const float inv = rsqrtf(var + 1e-5f);
            for (int i = 0; i < DIM; ++i)
                sQin[lane][i] = (frow[i] - mu) * inv * ln_g[i] + ln_b[i];
            bv = bidx[gr];
        } else {
            for (int i = 0; i < DIM; ++i) sQin[lane][i] = 0.f;
        }
        sBidx[lane] = bv;
    }
    __syncthreads();
    if (lane == 0) {
        int lo = 0x7fffffff, hi = -0x7fffffff;
        for (int i = 0; i < 16; ++i) {
            const int v = sBidx[i];
            if (v >= 0) { lo = min(lo, v); hi = max(hi, v); }
        }
        sRange[0] = lo; sRange[1] = hi;
    }
    __syncthreads();
    const int blo = sRange[0], bhi = sRange[1];
    if (blo > bhi) return;   // wave-uniform

    // ---- Q = LN(F) @ Wq^T + bq  (batch independent), stored bf16 ----
    for (int nt = 0; nt < DIM / 16; ++nt) {
        v8f acc = zero8();
        for (int kt = 0; kt < DIM; kt += 32) {
            BF16x16 A, Bf;
#pragma unroll
            for (int s = 0; s < 8; ++s) {
                const int ka = kt + a_kbase(s, half);
                A.h[2*s]   = (__bf16)sQin[ln][ka];
                A.h[2*s+1] = (__bf16)sQin[ln][ka + 1];
                const int kb = kt + b_kbase(s, half);
                Bf.u[s] = *(const unsigned*)(Wqb + (long)(nt * 16 + ln) * DIM + kb);
            }
            acc = wmma_bf16(A, Bf, acc);
        }
        const float bqv = bq[nt * 16 + ln];
#pragma unroll
        for (int r = 0; r < 8; ++r)
            sQ[half * 8 + r][nt * 16 + ln] = (__bf16)(acc[r] + bqv);
    }
    __syncthreads();

    for (int b = blo; b <= bhi; ++b) {
        for (int row = 0; row < 16; ++row)
            for (int i = lane; i < DIM; i += 32) sO[row][i] = 0.f;
        __syncthreads();

        const __bf16* Kb = Kw + (long)b * MKEYS * DIM;
        const __bf16* Vb = Vw + (long)b * MKEYS * DIM;

        for (int h = 0; h < NHEAD; ++h) {
            const int dcol0 = h * DH;
            float mrun = -1e30f, srun = 0.f;   // per-row stats (lane<16 owns row)

            // A-fragment of Q for this head is chunk-invariant: hoist it
            BF16x16 Aq;
#pragma unroll
            for (int s = 0; s < 8; ++s)
                Aq.u[s] = *(const unsigned*)&sQ[ln][dcol0 + a_kbase(s, half)];

            // kick off async DMA of the first K/V chunk for this head
            stage_async_chunk(Kb + dcol0, sK[0], lane);
            stage_async_chunk(Vb + dcol0, sV[0], lane);

            int buf = 0;
            for (int c = 0; c < MKEYS; c += CHUNK, buf ^= 1) {
                const bool pre = (c + CHUNK) < MKEYS;
                if (pre) {   // prefetch next chunk into the other buffer
                    stage_async_chunk(Kb + (long)(c + CHUNK) * DIM + dcol0, sK[buf ^ 1], lane);
                    stage_async_chunk(Vb + (long)(c + CHUNK) * DIM + dcol0, sV[buf ^ 1], lane);
                    asm volatile("s_wait_asynccnt 0x20" ::: "memory"); // current 32 done
                } else {
                    asm volatile("s_wait_asynccnt 0x0" ::: "memory");
                }
                __syncthreads();

                // ---- scores S = Q_h @ K_h^T * scale (one WMMA per 16-key tile)
                for (int ntk = 0; ntk < 8; ++ntk) {
                    BF16x16 Bf;
#pragma unroll
                    for (int s = 0; s < 8; ++s)
                        Bf.u[s] = *(const unsigned*)&sK[buf][ntk * 16 + ln][b_kbase(s, half)];
                    v8f acc = wmma_bf16(Aq, Bf, zero8());
#pragma unroll
                    for (int r = 0; r < 8; ++r)
                        sS[half * 8 + r][ntk * 16 + ln] = acc[r] * SCALE;
                }
                __syncthreads();

                // ---- online softmax stats + probs + rescale O slice ----
                if (lane < 16) {
                    float cmax = -1e30f;
                    for (int j = 0; j < CHUNK; ++j) cmax = fmaxf(cmax, sS[lane][j]);
                    const float mnew  = fmaxf(mrun, cmax);
                    const float scale = __expf(mrun - mnew);
                    float ssum = 0.f;
                    for (int j = 0; j < CHUNK; ++j) {
                        const float p = __expf(sS[lane][j] - mnew);
                        ssum += p;
                        sP[lane][j] = (__bf16)p;
                    }
                    srun = srun * scale + ssum;
                    mrun = mnew;
                    if (scale != 1.f)
                        for (int j = 0; j < DH; ++j) sO[lane][dcol0 + j] *= scale;
                }
                __syncthreads();

                // ---- O_h += P @ V_h  (two 16-wide D tiles per head) ----
                for (int ntd = 0; ntd < 2; ++ntd) {
                    const int dc = dcol0 + ntd * 16;
                    v8f acc;
#pragma unroll
                    for (int r = 0; r < 8; ++r) acc[r] = sO[half * 8 + r][dc + ln];
                    for (int kt = 0; kt < 4; ++kt) {
                        BF16x16 A, Bf;
#pragma unroll
                        for (int s = 0; s < 8; ++s) {
                            A.u[s] = *(const unsigned*)&sP[ln][kt * 32 + a_kbase(s, half)];
                            const int kb = kt * 32 + b_kbase(s, half);
                            Bf.h[2*s]   = sV[buf][kb]    [ntd * 16 + ln];
                            Bf.h[2*s+1] = sV[buf][kb + 1][ntd * 16 + ln];
                        }
                        acc = wmma_bf16(A, Bf, acc);
                    }
#pragma unroll
                    for (int r = 0; r < 8; ++r) sO[half * 8 + r][dc + ln] = acc[r];
                }
                __syncthreads();
            } // key chunks

            if (lane < 16) {
                const float inv = 1.f / srun;
                for (int j = 0; j < DH; ++j) sO[lane][dcol0 + j] *= inv;
            }
            __syncthreads();
        } // heads

        // ---- attn = O @ Wo^T + bo ; sR = q_in + attn ----
        for (int nt = 0; nt < DIM / 16; ++nt) {
            v8f acc = zero8();
            for (int kt = 0; kt < DIM; kt += 32) {
                BF16x16 A, Bf;
#pragma unroll
                for (int s = 0; s < 8; ++s) {
                    const int ka = kt + a_kbase(s, half);
                    A.h[2*s]   = (__bf16)sO[ln][ka];
                    A.h[2*s+1] = (__bf16)sO[ln][ka + 1];
                    const int kb = kt + b_kbase(s, half);
                    Bf.u[s] = *(const unsigned*)(Wob + (long)(nt * 16 + ln) * DIM + kb);
                }
                acc = wmma_bf16(A, Bf, acc);
            }
            const float bov = bo[nt * 16 + ln];
#pragma unroll
            for (int r = 0; r < 8; ++r) {
                const int m = half * 8 + r, ncol = nt * 16 + ln;
                sR[m][ncol] = sQin[m][ncol] + acc[r] + bov;
            }
        }
        __syncthreads();

        // ---- final LayerNorm + store rows of this batch ----
        if (lane < 16) {
            const int gr = r0 + lane;
            if (gr < nrows && sBidx[lane] == b) {
                float mu = 0.f;
                for (int i = 0; i < DIM; ++i) mu += sR[lane][i];
                mu *= (1.f / DIM);
                float var = 0.f;
                for (int i = 0; i < DIM; ++i) { float d = sR[lane][i] - mu; var += d * d; }
                var *= (1.f / DIM);
                const float inv = rsqrtf(var + 1e-5f);
                float* orow = out + (long)gr * DIM;
                for (int i = 0; i < DIM; ++i)
                    orow[i] = (sR[lane][i] - mu) * inv * ln_g[i] + ln_b[i];
            }
        }
        __syncthreads();
    } // batch loop
}

// ============================================================================
extern "C" void kernel_launch(void* const* d_in, const int* in_sizes, int n_in,
                              void* d_out, int out_size, void* d_ws, size_t ws_size,
                              hipStream_t stream)
{
    const float* F         = (const float*)d_in[0];
    const int*   batch_idx = (const int*)  d_in[1];
    const float* y         = (const float*)d_in[2];
    const float* Wq        = (const float*)d_in[3];
    const float* bq        = (const float*)d_in[4];
    const float* Wk        = (const float*)d_in[5];
    const float* bk        = (const float*)d_in[6];
    const float* Wv        = (const float*)d_in[7];
    const float* bv        = (const float*)d_in[8];
    const float* Wo        = (const float*)d_in[9];
    const float* bo        = (const float*)d_in[10];
    const float* ln_g      = (const float*)d_in[11];
    const float* ln_b      = (const float*)d_in[12];
    const int nrows = in_sizes[1];   // N

    __bf16* Kw  = (__bf16*)d_ws;
    __bf16* Vw  = Kw + (size_t)NBATCH * MKEYS * DIM;
    __bf16* Wqb = Vw + (size_t)NBATCH * MKEYS * DIM;
    __bf16* Wob = Wqb + (size_t)DIM * DIM;

    wconv_kernel<<<(DIM * DIM + 255) / 256, 256, 0, stream>>>(Wq, Wo, Wqb, Wob);

    dim3 g1(NBATCH * MKEYS / 16, DIM / 16, 2);
    kv_proj_kernel<<<g1, 32, 0, stream>>>(y, Wk, bk, Wv, bv, Kw, Vw);

    const int ntiles = (nrows + 15) / 16;
    attn_kernel<<<ntiles, 32, 0, stream>>>(F, batch_idx, Wqb, bq, Wob, bo,
                                           ln_g, ln_b, Kw, Vw,
                                           (float*)d_out, nrows);
}
